// SelfAttention_24017457119822
// MI455X (gfx1250) — matrix operand order repaired and checked by reading, not compiled, not run on hardware
//
#include <hip/hip_runtime.h>
#include <hip/hip_bf16.h>

typedef __bf16 bf16_t;
typedef __attribute__((ext_vector_type(16))) __bf16 v16bf;
typedef __attribute__((ext_vector_type(8)))  __bf16 v8bf;
typedef __attribute__((ext_vector_type(4)))  __bf16 v4bf;
typedef __attribute__((ext_vector_type(8)))  float  v8f;
typedef __attribute__((__vector_size__(16))) int    i32x4;

#define EMBED    1024
#define HEADS    16
#define HEAD_DIM 64
#define NBATCH   4
#define SEQ      2048
#define MROWS    (NBATCH * SEQ)   // 8192

// Async global->LDS (CDNA5) if the toolchain exposes it; else VGPR staging.
#ifndef USE_ASYNC_LDS
#if defined(__has_builtin)
#if __has_builtin(__builtin_amdgcn_global_load_async_to_lds_b128) && \
    __has_builtin(__builtin_amdgcn_s_wait_asynccnt)
#define USE_ASYNC_LDS 1
#else
#define USE_ASYNC_LDS 0
#endif
#else
#define USE_ASYNC_LDS 0
#endif
#endif

#if USE_ASYNC_LDS
// Param 0 confirmed by compiler diagnostic: AS1 pointer to int32x4.
#define ASYNC_G2L_B128(gsrc, ldst)                                          \
    __builtin_amdgcn_global_load_async_to_lds_b128(                         \
        (__attribute__((address_space(1))) i32x4*)(i32x4*)(gsrc),           \
        (__attribute__((address_space(3))) i32x4*)(i32x4*)(ldst), 0, 0)
#endif

// ---------------------------------------------------------------------------
__device__ __forceinline__ v8f wmma_bf16(v16bf a, v16bf b, v8f c) {
    return __builtin_amdgcn_wmma_f32_16x16x32_bf16(
        false, a, false, b, (short)0, c, false, false);
}

__device__ __forceinline__ v8f v8f_zero() {
    return (v8f){0.f, 0.f, 0.f, 0.f, 0.f, 0.f, 0.f, 0.f};
}

// A-fragment (16x32): lane group g holds K = {g*16..g*16+15} in elems 0..15
// (contiguous, matching the B fragments).
__device__ __forceinline__ v16bf afrag_bf16(const bf16_t* row, int g) {
    v8bf lo = *(const v8bf*)(row + g * 16);
    v8bf hi = *(const v8bf*)(row + g * 16 + 8);
    v16bf r;
#pragma unroll
    for (int i = 0; i < 8; i++) { r[i] = lo[i]; r[i + 8] = hi[i]; }
    return r;
}

// half-wave pair exchange (lane <-> lane^16) via LDS swizzle SWAPX16
__device__ __forceinline__ float swap16(float v) {
    int i = __builtin_bit_cast(int, v);
    i = __builtin_amdgcn_ds_swizzle(i, 0x401f);  // xor=0x10, and=0x1f
    return __builtin_bit_cast(float, i);
}

// ---------------------------------------------------------------------------
// Kernel 0: f32 -> bf16 conversion of x and the four weight matrices.
// ---------------------------------------------------------------------------
__global__ __launch_bounds__(256) void attn_cvt_kernel(
    const float* __restrict__ x,  const float* __restrict__ Wq,
    const float* __restrict__ Wk, const float* __restrict__ Wv,
    const float* __restrict__ Wo,
    bf16_t* __restrict__ xb,  bf16_t* __restrict__ wqb,
    bf16_t* __restrict__ wkb, bf16_t* __restrict__ wvb,
    bf16_t* __restrict__ wob)
{
    const float* src; bf16_t* dst; int count;
    switch (blockIdx.y) {
        case 0:  src = x;  dst = xb;  count = MROWS * EMBED; break;
        case 1:  src = Wq; dst = wqb; count = EMBED * EMBED; break;
        case 2:  src = Wk; dst = wkb; count = EMBED * EMBED; break;
        case 3:  src = Wv; dst = wvb; count = EMBED * EMBED; break;
        default: src = Wo; dst = wob; count = EMBED * EMBED; break;
    }
    int n4 = count >> 2;
    for (int i = blockIdx.x * blockDim.x + threadIdx.x; i < n4;
         i += gridDim.x * blockDim.x) {
        float4 v = ((const float4*)src)[i];
        v4bf o;
        o[0] = (bf16_t)v.x; o[1] = (bf16_t)v.y;
        o[2] = (bf16_t)v.z; o[3] = (bf16_t)v.w;
        ((v4bf*)dst)[i] = o;
    }
}

// ---------------------------------------------------------------------------
// Kernel 1: fused Q/K/V projection, bf16 in / bf16 out, 32x64 tile per wave.
// K-loop unrolled by 2 with ping-pong register sets (no copy movs).
// which=0 -> qbuf (x 1/sqrt(E)), 1 -> kbuf, 2 -> vt.
// ---------------------------------------------------------------------------
__global__ __launch_bounds__(32) void attn_proj_kernel(
    const bf16_t* __restrict__ xb,
    const bf16_t* __restrict__ wqb, const bf16_t* __restrict__ wkb,
    const bf16_t* __restrict__ wvb,
    bf16_t* __restrict__ qbuf, bf16_t* __restrict__ kbuf,
    bf16_t* __restrict__ vtbuf)
{
    const int lane = threadIdx.x & 31;
    const int g = lane >> 4, ln = lane & 15;
    const int mBase = blockIdx.x * 32;
    const int eBase = blockIdx.y * 64;
    const int which = blockIdx.z;
    const bf16_t* W = (which == 0) ? wqb : ((which == 1) ? wkb : wvb);

    const bf16_t* xr[2];
    xr[0] = xb + (size_t)(mBase + ln) * EMBED;
    xr[1] = xb + (size_t)(mBase + 16 + ln) * EMBED;
    const bf16_t* wr[4];
#pragma unroll
    for (int t = 0; t < 4; t++)
        wr[t] = W + (size_t)(eBase + t * 16 + ln) * EMBED + g * 16;

    v8f acc[2][4];
#pragma unroll
    for (int m = 0; m < 2; m++)
#pragma unroll
        for (int t = 0; t < 4; t++) acc[m][t] = v8f_zero();

    v16bf a0[2], b0[4], a1[2], b1[4];
#pragma unroll
    for (int m = 0; m < 2; m++) a0[m] = afrag_bf16(xr[m], g);
#pragma unroll
    for (int t = 0; t < 4; t++) b0[t] = *(const v16bf*)(wr[t]);

    for (int k0 = 0; k0 < EMBED; k0 += 64) {
        const int kA = k0 + 32;                       // always < EMBED
#pragma unroll
        for (int m = 0; m < 2; m++) a1[m] = afrag_bf16(xr[m] + kA, g);
#pragma unroll
        for (int t = 0; t < 4; t++) b1[t] = *(const v16bf*)(wr[t] + kA);
        if (k0 + 256 < EMBED) {
            __builtin_prefetch(xr[0] + k0 + 256, 0, 3);
            __builtin_prefetch(wr[0] + k0 + 256, 0, 3);
        }
#pragma unroll
        for (int m = 0; m < 2; m++)
#pragma unroll
            for (int t = 0; t < 4; t++)
                acc[m][t] = wmma_bf16(a0[m], b0[t], acc[m][t]);

        const int kB = (k0 + 64 < EMBED) ? (k0 + 64) : k0;  // last: dummy reload
#pragma unroll
        for (int m = 0; m < 2; m++) a0[m] = afrag_bf16(xr[m] + kB, g);
#pragma unroll
        for (int t = 0; t < 4; t++) b0[t] = *(const v16bf*)(wr[t] + kB);
#pragma unroll
        for (int m = 0; m < 2; m++)
#pragma unroll
            for (int t = 0; t < 4; t++)
                acc[m][t] = wmma_bf16(a1[m], b1[t], acc[m][t]);
    }

    const float osc = (which == 0) ? 0.03125f : 1.0f;  // fold 1/sqrt(EMBED) into Q
#pragma unroll
    for (int m2 = 0; m2 < 2; m2++) {
#pragma unroll
        for (int t = 0; t < 4; t++) {
#pragma unroll
            for (int r = 0; r < 8; r++) {
                int m = mBase + m2 * 16 + r + 8 * g;
                int e = eBase + t * 16 + ln;
                int n = m >> 11, s = m & (SEQ - 1);
                int h = e >> 6, d = e & 63;
                bf16_t val = (bf16_t)(acc[m2][t][r] * osc);
                if (which == 2)
                    vtbuf[((size_t)(n * HEADS + h) * HEAD_DIM + d) * SEQ + s] = val;
                else if (which == 1)
                    kbuf[((size_t)(n * HEADS + h) * SEQ + s) * HEAD_DIM + d] = val;
                else
                    qbuf[((size_t)(n * HEADS + h) * SEQ + s) * HEAD_DIM + d] = val;
            }
        }
    }
}

// ---------------------------------------------------------------------------
// Kernel 2: flash attention.  4 waves/block, same (n,h); waves cooperatively
// stage each 32-key K/V tile into LDS (async-to-LDS if available), each wave
// owns one 16-row q-tile.  Softmax row stats via ds_swizzle half-wave split.
// ---------------------------------------------------------------------------
__global__ __launch_bounds__(128) void attn_flash_kernel(
    const bf16_t* __restrict__ qbuf, const bf16_t* __restrict__ kbuf,
    const bf16_t* __restrict__ vtbuf, bf16_t* __restrict__ attb)
{
    __shared__ bf16_t sK[32][72];      // 32 keys x 64 d  (+8 pad)
    __shared__ bf16_t sV[64][40];      // 64 d  x 32 keys (+8 pad)
    __shared__ float  sS[4][16][33];   // per-wave score tile
    __shared__ float  sStat[4][16];    // per-wave row broadcast

    const int tid  = threadIdx.x;
    const int wid  = tid >> 5;
    const int lane = tid & 31;
    const int g = lane >> 4, ln = lane & 15;
    const int qt = blockIdx.x * 4 + wid;
    const int h = blockIdx.y, n = blockIdx.z;
    const int nh = n * HEADS + h;
    const size_t qkBase = (size_t)nh * SEQ * HEAD_DIM;
    const size_t vtBase = (size_t)nh * HEAD_DIM * SEQ;

    // per-wave Q fragments (scale already folded in at projection time)
    const bf16_t* qrow = qbuf + qkBase + (size_t)(qt * 16 + ln) * HEAD_DIM;
    v16bf qa[2];
#pragma unroll
    for (int c = 0; c < 2; c++) qa[c] = afrag_bf16(qrow + c * 32, g);

    v8f o[4];
#pragma unroll
    for (int t = 0; t < 4; t++) o[t] = v8f_zero();
    float mrow = -__builtin_inff(), lrow = 0.f;

    for (int kb = 0; kb < SEQ; kb += 32) {
        // ---- cooperative staging of K (4KB) and V (4KB) tiles ----
#pragma unroll
        for (int q = 0; q < 2; q++) {
            int c = tid * 2 + q;                    // 0..255 16-byte chunks
            int kr = c >> 3, kc = (c & 7) * 8;
            int vr = c >> 2, vc = (c & 3) * 8;
            const bf16_t* ksrc = kbuf + qkBase + (size_t)(kb + kr) * HEAD_DIM + kc;
            const bf16_t* vsrc = vtbuf + vtBase + (size_t)vr * SEQ + kb + vc;
#if USE_ASYNC_LDS
            ASYNC_G2L_B128(ksrc, &sK[kr][kc]);
            ASYNC_G2L_B128(vsrc, &sV[vr][vc]);
#else
            *(v8bf*)&sK[kr][kc] = *(const v8bf*)ksrc;
            *(v8bf*)&sV[vr][vc] = *(const v8bf*)vsrc;
#endif
            if (kb + 32 < SEQ) {
                __builtin_prefetch(ksrc + 32 * HEAD_DIM, 0, 3);
                __builtin_prefetch(vsrc + 32, 0, 3);
            }
        }
#if USE_ASYNC_LDS
        __builtin_amdgcn_s_wait_asynccnt(0);
#endif
        __syncthreads();

        // ---- S = Q K^T for 32 keys (two 16-key column tiles) ----
        v8f sc[2];
#pragma unroll
        for (int t = 0; t < 2; t++) {
            sc[t] = v8f_zero();
#pragma unroll
            for (int c = 0; c < 2; c++) {
                v16bf b = *(const v16bf*)(&sK[t * 16 + ln][c * 32 + g * 16]);
                sc[t] = wmma_bf16(qa[c], b, sc[t]);
            }
        }
#pragma unroll
        for (int t = 0; t < 2; t++)
#pragma unroll
            for (int r = 0; r < 8; r++)
                sS[wid][r + 8 * g][t * 16 + ln] = sc[t][r];
        asm volatile("" ::: "memory");  // DS ops are in-order within a wave

        // ---- online softmax: lane (ln,g) owns cols [16g,16g+16) of row ln ----
        float sv[16];
#pragma unroll
        for (int i = 0; i < 16; i++) sv[i] = sS[wid][ln][g * 16 + i];
        float mt = sv[0];
#pragma unroll
        for (int i = 1; i < 16; i++) mt = fmaxf(mt, sv[i]);
        mt = fmaxf(mt, swap16(mt));                 // combine half-waves
        float mnew = fmaxf(mrow, mt);
        float corr = __expf(mrow - mnew);
        float p[16];
        float rs = 0.f;
#pragma unroll
        for (int i = 0; i < 16; i++) { p[i] = __expf(sv[i] - mnew); rs += p[i]; }
        rs += swap16(rs);
        lrow = lrow * corr + rs;
        mrow = mnew;
        sStat[wid][ln] = corr;                      // dup write, same value
        asm volatile("" ::: "memory");

        // ---- rescale O ----
#pragma unroll
        for (int r = 0; r < 8; r++) {
            float cr = sStat[wid][r + 8 * g];
#pragma unroll
            for (int t = 0; t < 4; t++) o[t][r] *= cr;
        }

        // ---- build P A-fragment from own p values (no exchange, no LDS tile) ----
        // lane group g needs cols {16g..16g+15} = own p[0..15] (contiguous).
        v16bf pa;
#pragma unroll
        for (int i = 0; i < 16; i++) {
            pa[i] = (bf16_t)p[i];
        }

        // ---- O += P V ----
#pragma unroll
        for (int t = 0; t < 4; t++) {
            v16bf b = *(const v16bf*)(&sV[t * 16 + ln][g * 16]);
            o[t] = wmma_bf16(pa, b, o[t]);
        }
        __syncthreads();   // protect sK/sV before next staging round
    }

    // ---- epilogue: O /= l, store bf16 to attb[n, s, h*64 + d] ----
    sStat[wid][ln] = 1.0f / lrow;
    asm volatile("" ::: "memory");
#pragma unroll
    for (int r = 0; r < 8; r++) {
        float linv = sStat[wid][r + 8 * g];
        int s = qt * 16 + r + 8 * g;
        size_t rowoff = ((size_t)n * SEQ + s) * EMBED + h * HEAD_DIM;
#pragma unroll
        for (int t = 0; t < 4; t++)
            attb[rowoff + t * 16 + ln] = (bf16_t)(o[t][r] * linv);
    }
}

// ---------------------------------------------------------------------------
// Kernel 3: output projection  out = attb @ Wo^T + bo  (f32 out), 32x64 tile,
// same ping-pong K-loop as the projection kernel.
// ---------------------------------------------------------------------------
__global__ __launch_bounds__(32) void attn_outproj_kernel(
    const bf16_t* __restrict__ attb, const bf16_t* __restrict__ wob,
    const float* __restrict__ bo, float* __restrict__ out)
{
    const int lane = threadIdx.x & 31;
    const int g = lane >> 4, ln = lane & 15;
    const int mBase = blockIdx.x * 32;
    const int eBase = blockIdx.y * 64;

    const bf16_t* ar[2];
    ar[0] = attb + (size_t)(mBase + ln) * EMBED;
    ar[1] = attb + (size_t)(mBase + 16 + ln) * EMBED;
    const bf16_t* wr[4];
#pragma unroll
    for (int t = 0; t < 4; t++)
        wr[t] = wob + (size_t)(eBase + t * 16 + ln) * EMBED + g * 16;

    v8f acc[2][4];
#pragma unroll
    for (int m = 0; m < 2; m++)
#pragma unroll
        for (int t = 0; t < 4; t++) acc[m][t] = v8f_zero();

    v16bf a0[2], b0[4], a1[2], b1[4];
#pragma unroll
    for (int m = 0; m < 2; m++) a0[m] = afrag_bf16(ar[m], g);
#pragma unroll
    for (int t = 0; t < 4; t++) b0[t] = *(const v16bf*)(wr[t]);

    for (int k0 = 0; k0 < EMBED; k0 += 64) {
        const int kA = k0 + 32;                       // always < EMBED
#pragma unroll
        for (int m = 0; m < 2; m++) a1[m] = afrag_bf16(ar[m] + kA, g);
#pragma unroll
        for (int t = 0; t < 4; t++) b1[t] = *(const v16bf*)(wr[t] + kA);
        if (k0 + 256 < EMBED) {
            __builtin_prefetch(ar[0] + k0 + 256, 0, 3);
            __builtin_prefetch(wr[0] + k0 + 256, 0, 3);
        }
#pragma unroll
        for (int m = 0; m < 2; m++)
#pragma unroll
            for (int t = 0; t < 4; t++)
                acc[m][t] = wmma_bf16(a0[m], b0[t], acc[m][t]);

        const int kB = (k0 + 64 < EMBED) ? (k0 + 64) : k0;  // last: dummy reload
#pragma unroll
        for (int m = 0; m < 2; m++) a0[m] = afrag_bf16(ar[m] + kB, g);
#pragma unroll
        for (int t = 0; t < 4; t++) b0[t] = *(const v16bf*)(wr[t] + kB);
#pragma unroll
        for (int m = 0; m < 2; m++)
#pragma unroll
            for (int t = 0; t < 4; t++)
                acc[m][t] = wmma_bf16(a1[m], b1[t], acc[m][t]);
    }

#pragma unroll
    for (int m2 = 0; m2 < 2; m2++)
#pragma unroll
        for (int t = 0; t < 4; t++)
#pragma unroll
            for (int r = 0; r < 8; r++) {
                int m = mBase + m2 * 16 + r + 8 * g;
                int e = eBase + t * 16 + ln;
                out[(size_t)m * EMBED + e] = acc[m2][t][r] + bo[e];
            }
}

// ---------------------------------------------------------------------------
extern "C" void kernel_launch(void* const* d_in, const int* in_sizes, int n_in,
                              void* d_out, int out_size, void* d_ws, size_t ws_size,
                              hipStream_t stream) {
    const float* x  = (const float*)d_in[0];
    const float* Wq = (const float*)d_in[1];
    const float* Wk = (const float*)d_in[2];
    const float* Wv = (const float*)d_in[3];
    const float* Wo = (const float*)d_in[4];
    const float* bo = (const float*)d_in[5];
    float* out = (float*)d_out;

    const size_t X_BYTES = (size_t)MROWS * EMBED * sizeof(bf16_t);  // 16 MB
    const size_t W_BYTES = (size_t)EMBED * EMBED * sizeof(bf16_t);  //  2 MB
    char* ws = (char*)d_ws;
    bf16_t* xb    = (bf16_t*)(ws);
    bf16_t* wqb   = (bf16_t*)(ws + X_BYTES);
    bf16_t* wkb   = (bf16_t*)(ws + X_BYTES + 1 * W_BYTES);
    bf16_t* wvb   = (bf16_t*)(ws + X_BYTES + 2 * W_BYTES);
    bf16_t* wob   = (bf16_t*)(ws + X_BYTES + 3 * W_BYTES);
    bf16_t* qbuf  = (bf16_t*)(ws + X_BYTES + 4 * W_BYTES);
    bf16_t* kbuf  = (bf16_t*)(ws + 2 * X_BYTES + 4 * W_BYTES);
    bf16_t* vtbuf = (bf16_t*)(ws + 3 * X_BYTES + 4 * W_BYTES);
    bf16_t* attb  = (bf16_t*)(ws + 4 * X_BYTES + 4 * W_BYTES);

    attn_cvt_kernel<<<dim3(512, 5), 256, 0, stream>>>(
        x, Wq, Wk, Wv, Wo, xb, wqb, wkb, wvb, wob);
    attn_proj_kernel<<<dim3(MROWS / 32, EMBED / 64, 3), 32, 0, stream>>>(
        xb, wqb, wkb, wvb, qbuf, kbuf, vtbuf);
    attn_flash_kernel<<<dim3(SEQ / 16 / 4, HEADS, NBATCH), 128, 0, stream>>>(
        qbuf, kbuf, vtbuf, attb);
    attn_outproj_kernel<<<dim3(MROWS / 32, EMBED / 64), 32, 0, stream>>>(
        attb, wob, bo, out);
}
